// LSTM_45578192945278
// MI455X (gfx1250) — matrix operand order repaired and checked by reading, not compiled, not run on hardware
//
#include <hip/hip_runtime.h>
#include <hip/hip_bf16.h>

// ---------------- problem constants ----------------
#define H1   60
#define H2   65
#define IN_  32
#define Bsz  1024
#define Tsz  512
#define G1N  240   // 4*H1  -> exactly 15 WMMA N-tiles
#define G2N  272   // 4*H2=260 padded to 17 N-tiles
#define K1P  64    // H1 padded to 2 k-steps of 32
#define K2P  96    // H2 padded to 3 k-steps of 32

typedef __attribute__((ext_vector_type(16))) __bf16 v16bf;
typedef __attribute__((ext_vector_type(8)))  float  v8f;

union FragU { v16bf v; unsigned short u[16]; };

// ---------------- LDS layout (bytes) ----------------
constexpr int OFF_G1  = 0;                       // [16][G1N] f32
constexpr int OFF_G2  = OFF_G1 + 16 * G1N * 4;   // [16][G2N] f32
constexpr int OFF_C1  = OFF_G2 + 16 * G2N * 4;   // [16][H1]  f32
constexpr int OFF_C2  = OFF_C1 + 16 * H1 * 4;    // [16][H2]  f32
constexpr int OFF_B1  = OFF_C2 + 16 * H2 * 4;    // [G1N]     f32 (b_ih1+b_hh1)
constexpr int OFF_B2  = OFF_B1 + G1N * 4;        // [260]     f32
constexpr int OFF_WI1 = OFF_B2 + 260 * 4;        // [32 ][G1N] bf16 (W_ih1^T)
constexpr int OFF_WH1 = OFF_WI1 + 32  * G1N * 2; // [K1P][G1N] bf16 (W_hh1^T, zero pad)
constexpr int OFF_WI2 = OFF_WH1 + K1P * G1N * 2; // [K1P][G2N] bf16 (W_ih2^T)
constexpr int OFF_WH2 = OFF_WI2 + K1P * G2N * 2; // [K2P][G2N] bf16 (W_hh2^T)
constexpr int OFF_XT  = OFF_WH2 + K2P * G2N * 2; // [16][IN_]  bf16 (x_t tile)
constexpr int OFF_H1B = OFF_XT  + 16 * IN_ * 2;  // [16][K1P]  bf16 (h1 state)
constexpr int OFF_H2B = OFF_H1B + 16 * K1P * 2;  // [16][K2P]  bf16 (h2 state)
constexpr int SMEM_BYTES = OFF_H2B + 16 * K2P * 2;  // ~178 KB < 320 KB/WGP

// ---------------- helpers ----------------
__device__ __forceinline__ unsigned short f2bf(float f) {
  union { float f; unsigned int u; } x; x.f = f;
  unsigned int r = x.u + 0x7FFFu + ((x.u >> 16) & 1u);  // RNE
  return (unsigned short)(r >> 16);
}
__device__ __forceinline__ float sigf(float v) { return 1.0f / (1.0f + __expf(-v)); }
__device__ __forceinline__ float tanf_(float v) { return 1.0f - 2.0f / (__expf(2.0f * v) + 1.0f); }

// A fragment: 16x32 bf16 tile from LDS row-major [M][K], contiguous k order (same as gatherB):
// lane&15 = M ; K = e + ((lane>>4)<<4)
__device__ __forceinline__ FragU gatherA(const unsigned short* base, int kstride,
                                         int kbase, int lane) {
  FragU f;
  const unsigned short* row = base + (lane & 15) * kstride + kbase + (((lane >> 4) & 1) << 4);
#pragma unroll
  for (int e = 0; e < 16; ++e) f.u[e] = row[e];
  return f;
}

// B fragment: 32x16 bf16 tile from LDS row-major [K][N]:
// lane&15 = N ; K = (lane&16) + e
__device__ __forceinline__ FragU gatherB(const unsigned short* base, int nstride,
                                         int kbase, int n0, int lane) {
  FragU f;
  const unsigned short* p = base + (kbase + (lane & 16)) * nstride + n0 + (lane & 15);
#pragma unroll
  for (int e = 0; e < 16; ++e) f.u[e] = p[e * nstride];
  return f;
}

// C/D 16x16 f32: VGPR r -> M = r + ((lane>>4)<<3), N = lane&15
__device__ __forceinline__ void storeC(float* g, int nstride, int n0, v8f acc, int lane) {
  float* p = g + (((lane >> 4) & 1) << 3) * nstride + n0 + (lane & 15);
#pragma unroll
  for (int r = 0; r < 8; ++r) p[r * nstride] = acc[r];
}

__device__ __forceinline__ v8f wmma_bf16(FragU a, FragU b, v8f c) {
  return __builtin_amdgcn_wmma_f32_16x16x32_bf16(false, a.v, false, b.v,
                                                 (short)0, c, false, false);
}

// ---------------- fused 2-layer LSTM ----------------
// One workgroup = one 16-row batch tile, 8 waves, full T loop, both layers.
__global__ __launch_bounds__(256) void lstm2_fused(
    const float* __restrict__ x,
    const float* __restrict__ Wih1, const float* __restrict__ Whh1,
    const float* __restrict__ bih1, const float* __restrict__ bhh1,
    const float* __restrict__ Wih2, const float* __restrict__ Whh2,
    const float* __restrict__ bih2, const float* __restrict__ bhh2,
    float* __restrict__ h2_final /* [B][H2] in workspace */) {
  extern __shared__ char smem[];
  float*          g1    = (float*)(smem + OFF_G1);
  float*          g2    = (float*)(smem + OFF_G2);
  float*          c1    = (float*)(smem + OFF_C1);
  float*          c2    = (float*)(smem + OFF_C2);
  float*          bias1 = (float*)(smem + OFF_B1);
  float*          bias2 = (float*)(smem + OFF_B2);
  unsigned short* wi1   = (unsigned short*)(smem + OFF_WI1);
  unsigned short* wh1   = (unsigned short*)(smem + OFF_WH1);
  unsigned short* wi2   = (unsigned short*)(smem + OFF_WI2);
  unsigned short* wh2   = (unsigned short*)(smem + OFF_WH2);
  unsigned short* xt    = (unsigned short*)(smem + OFF_XT);
  unsigned short* h1b   = (unsigned short*)(smem + OFF_H1B);
  unsigned short* h2b   = (unsigned short*)(smem + OFF_H2B);

  const int tid  = threadIdx.x;
  const int lane = tid & 31;
  const int wave = tid >> 5;
  const int b0   = blockIdx.x * 16;

  // ---- one-time init: bf16 weight transposes (zero-padded), biases, state ----
  for (int i = tid; i < 32 * G1N; i += 256) {
    int k = i / G1N, n = i % G1N;
    wi1[i] = f2bf(Wih1[n * IN_ + k]);
  }
  for (int i = tid; i < K1P * G1N; i += 256) {
    int k = i / G1N, n = i % G1N;
    wh1[i] = (k < H1) ? f2bf(Whh1[n * H1 + k]) : (unsigned short)0;
  }
  for (int i = tid; i < K1P * G2N; i += 256) {
    int k = i / G2N, n = i % G2N;
    wi2[i] = (k < H1 && n < 4 * H2) ? f2bf(Wih2[n * H1 + k]) : (unsigned short)0;
  }
  for (int i = tid; i < K2P * G2N; i += 256) {
    int k = i / G2N, n = i % G2N;
    wh2[i] = (k < H2 && n < 4 * H2) ? f2bf(Whh2[n * H2 + k]) : (unsigned short)0;
  }
  for (int i = tid; i < G1N; i += 256)      bias1[i] = bih1[i] + bhh1[i];
  for (int i = tid; i < 4 * H2; i += 256)   bias2[i] = bih2[i] + bhh2[i];
  for (int i = tid; i < 16 * H1; i += 256)  c1[i] = 0.0f;
  for (int i = tid; i < 16 * H2; i += 256)  c2[i] = 0.0f;
  for (int i = tid; i < 16 * K1P; i += 256) h1b[i] = 0;
  for (int i = tid; i < 16 * K2P; i += 256) h2b[i] = 0;
  __syncthreads();

  // ---- hoist time-invariant weight (B) fragments into registers ----
  // L1: 15 N-tiles over 8 waves -> wave w owns {w, w+8} (wave 7 only one)
  const int nT1 = (wave < 7) ? 2 : 1;
  const int t1n0[2] = { wave * 16, (wave + 8) * 16 };
  FragU bW1[2][3];
#pragma unroll
  for (int i = 0; i < 2; ++i)
    if (i < nT1) {
      bW1[i][0] = gatherB(wi1, G1N, 0,  t1n0[i], lane);
      bW1[i][1] = gatherB(wh1, G1N, 0,  t1n0[i], lane);
      bW1[i][2] = gatherB(wh1, G1N, 32, t1n0[i], lane);
    }
  // L2: 17 N-tiles -> wave w owns {w, w+8}, wave 0 additionally tile 16
  const int nT2 = (wave == 0) ? 3 : 2;
  const int t2n0[3] = { wave * 16, (wave + 8) * 16, 16 * 16 };
  FragU bW2[3][5];
#pragma unroll
  for (int i = 0; i < 3; ++i)
    if (i < nT2) {
      bW2[i][0] = gatherB(wi2, G2N, 0,  t2n0[i], lane);
      bW2[i][1] = gatherB(wi2, G2N, 32, t2n0[i], lane);
      bW2[i][2] = gatherB(wh2, G2N, 0,  t2n0[i], lane);
      bW2[i][3] = gatherB(wh2, G2N, 32, t2n0[i], lane);
      bW2[i][4] = gatherB(wh2, G2N, 64, t2n0[i], lane);
    }

  // ---- sequential scan over time ----
  for (int t = 0; t < Tsz; ++t) {
    // Phase A: stage x_t tile (16x32) into LDS as bf16; prefetch x_{t+1}
    for (int i = tid; i < 16 * IN_; i += 256) {
      int m = i >> 5, ci = i & 31;
      xt[i] = f2bf(x[((size_t)(b0 + m) * Tsz + t) * IN_ + ci]);
    }
    if (t + 1 < Tsz && tid < 16)
      __builtin_prefetch(&x[((size_t)(b0 + tid) * Tsz + (t + 1)) * IN_], 0, 0);
    __syncthreads();

    // Phase B: layer-1 gates  g1 = x_t @ Wih1^T + h1 @ Whh1^T   (WMMA bf16)
    FragU aX = gatherA(xt, IN_, 0, lane);
    FragU aH1_0 = gatherA(h1b, K1P, 0, lane);
    FragU aH1_1 = gatherA(h1b, K1P, 32, lane);
#pragma unroll
    for (int i = 0; i < 2; ++i)
      if (i < nT1) {
        v8f acc = {};
        acc = wmma_bf16(aX,    bW1[i][0], acc);
        acc = wmma_bf16(aH1_0, bW1[i][1], acc);
        acc = wmma_bf16(aH1_1, bW1[i][2], acc);
        storeC(g1, G1N, t1n0[i], acc, lane);
      }
    __syncthreads();

    // Phase C: layer-1 pointwise gates, update c1/h1
    for (int i = tid; i < 16 * H1; i += 256) {
      int m = i / H1, j = i - m * H1;
      const float* gr = g1 + m * G1N;
      float gi = sigf (gr[j]          + bias1[j]);
      float gf = sigf (gr[H1 + j]     + bias1[H1 + j]);
      float gg = tanf_(gr[2 * H1 + j] + bias1[2 * H1 + j]);
      float go = sigf (gr[3 * H1 + j] + bias1[3 * H1 + j]);
      float c  = gf * c1[m * H1 + j] + gi * gg;
      c1[m * H1 + j] = c;
      h1b[m * K1P + j] = f2bf(go * tanf_(c));
    }
    __syncthreads();

    // Phase D: layer-2 gates  g2 = h1_t @ Wih2^T + h2 @ Whh2^T
    aH1_0 = gatherA(h1b, K1P, 0, lane);
    aH1_1 = gatherA(h1b, K1P, 32, lane);
    FragU aH2_0 = gatherA(h2b, K2P, 0, lane);
    FragU aH2_1 = gatherA(h2b, K2P, 32, lane);
    FragU aH2_2 = gatherA(h2b, K2P, 64, lane);
#pragma unroll
    for (int i = 0; i < 3; ++i)
      if (i < nT2) {
        v8f acc = {};
        acc = wmma_bf16(aH1_0, bW2[i][0], acc);
        acc = wmma_bf16(aH1_1, bW2[i][1], acc);
        acc = wmma_bf16(aH2_0, bW2[i][2], acc);
        acc = wmma_bf16(aH2_1, bW2[i][3], acc);
        acc = wmma_bf16(aH2_2, bW2[i][4], acc);
        storeC(g2, G2N, t2n0[i], acc, lane);
      }
    __syncthreads();

    // Phase E: layer-2 pointwise gates, update c2/h2; emit final h2
    for (int i = tid; i < 16 * H2; i += 256) {
      int m = i / H2, j = i - m * H2;
      const float* gr = g2 + m * G2N;
      float gi = sigf (gr[j]          + bias2[j]);
      float gf = sigf (gr[H2 + j]     + bias2[H2 + j]);
      float gg = tanf_(gr[2 * H2 + j] + bias2[2 * H2 + j]);
      float go = sigf (gr[3 * H2 + j] + bias2[3 * H2 + j]);
      float c  = gf * c2[m * H2 + j] + gi * gg;
      c2[m * H2 + j] = c;
      float h = go * tanf_(c);
      h2b[m * K2P + j] = f2bf(h);
      if (t == Tsz - 1) h2_final[(size_t)(b0 + m) * H2 + j] = h;
    }
    __syncthreads();
  }
}

// ---------------- tiny MLP head + softmax ----------------
__global__ __launch_bounds__(256) void lstm_head(
    const float* __restrict__ h2f, const float* __restrict__ W1,
    const float* __restrict__ b1, const float* __restrict__ W2,
    const float* __restrict__ b2, float* __restrict__ out) {
  int b = blockIdx.x * blockDim.x + threadIdx.x;
  if (b >= Bsz) return;
  const float* h = h2f + (size_t)b * H2;
  float e[16];
#pragma unroll
  for (int d = 0; d < 16; ++d) {
    float s = b1[d];
    for (int j = 0; j < H2; ++j) s += W1[d * H2 + j] * h[j];
    e[d] = s;
  }
  float l0 = b2[0], l1 = b2[1];
#pragma unroll
  for (int d = 0; d < 16; ++d) { l0 += W2[d] * e[d]; l1 += W2[16 + d] * e[d]; }
  float mx = fmaxf(l0, l1);
  float p0 = __expf(l0 - mx), p1 = __expf(l1 - mx);
  float inv = 1.0f / (p0 + p1);
  out[b * 2 + 0] = p0 * inv;
  out[b * 2 + 1] = p1 * inv;
#pragma unroll
  for (int d = 0; d < 16; ++d) out[Bsz * 2 + b * 16 + d] = e[d];
}

extern "C" void kernel_launch(void* const* d_in, const int* in_sizes, int n_in,
                              void* d_out, int out_size, void* d_ws, size_t ws_size,
                              hipStream_t stream) {
  (void)in_sizes; (void)n_in; (void)out_size; (void)ws_size;
  const float* x    = (const float*)d_in[0];
  const float* Wih1 = (const float*)d_in[1];
  const float* Whh1 = (const float*)d_in[2];
  const float* bih1 = (const float*)d_in[3];
  const float* bhh1 = (const float*)d_in[4];
  const float* Wih2 = (const float*)d_in[5];
  const float* Whh2 = (const float*)d_in[6];
  const float* bih2 = (const float*)d_in[7];
  const float* bhh2 = (const float*)d_in[8];
  const float* W1   = (const float*)d_in[9];
  const float* b1   = (const float*)d_in[10];
  const float* W2   = (const float*)d_in[11];
  const float* b2   = (const float*)d_in[12];

  float* h2f = (float*)d_ws;  // [B][H2] f32, fully written before read

  hipFuncSetAttribute(reinterpret_cast<const void*>(lstm2_fused),
                      hipFuncAttributeMaxDynamicSharedMemorySize, SMEM_BYTES);
  lstm2_fused<<<Bsz / 16, 256, SMEM_BYTES, stream>>>(
      x, Wih1, Whh1, bih1, bhh1, Wih2, Whh2, bih2, bhh2, h2f);
  lstm_head<<<(Bsz + 255) / 256, 256, 0, stream>>>(h2f, W1, b1, W2, b2,
                                                   (float*)d_out);
}
